// GroupARouter_78288663872327
// MI455X (gfx1250) — matrix operand order repaired and checked by reading, not compiled, not run on hardware
//
#include <hip/hip_runtime.h>
#include <math.h>

// Problem constants (from reference)
#define BB   8
#define NN   16384
#define DD   256
#define EE   8
#define HH   128
#define KSEL 4096   // tokens_per_expert = N*TOP_K/E = 16384*2/8
#define TPB  64     // 64 tokens = 4 tiles of 16 (unused; the fused GEMM block is TOKS_PER_BLOCK)

typedef __attribute__((ext_vector_type(16))) __bf16 v16bf;
typedef __attribute__((ext_vector_type(8)))  __bf16 v8bf;
typedef __attribute__((ext_vector_type(8)))  float  v8f;

// TDM descriptor vector types (per CDNA5_HIP.md probe)
typedef __attribute__((ext_vector_type(4))) unsigned tdm_g0_t;
typedef __attribute__((ext_vector_type(8))) int      tdm_g1_t;
typedef __attribute__((ext_vector_type(4))) int      tdm_g2_t;

#if defined(__HIP_DEVICE_COMPILE__) && __has_builtin(__builtin_amdgcn_tensor_load_to_lds) && \
    __has_builtin(__builtin_amdgcn_s_wait_tensorcnt)
#define HAVE_TDM 1
#else
#define HAVE_TDM 0
#endif

// Monotone-increasing mapping from float to unsigned (for descending radix select)
__device__ __forceinline__ unsigned fkey(float x) {
  unsigned u = __float_as_uint(x);
  return (u & 0x80000000u) ? ~u : (u | 0x80000000u);
}

// float -> bf16 bits, round-to-nearest-even
__device__ __forceinline__ unsigned short f2bf(float x) {
  unsigned u = __float_as_uint(x);
  u += 0x7FFFu + ((u >> 16) & 1u);
  return (unsigned short)(u >> 16);
}

// ---------------------------------------------------------------------------
// Kernel 0: prep — transpose W1 -> bf16 [H][D], pad W2 -> bf16 [16][H], zero sum
// ---------------------------------------------------------------------------
__global__ void prep_kernel(const float* __restrict__ W1, const float* __restrict__ W2,
                            unsigned short* __restrict__ W1t, unsigned short* __restrict__ W2t,
                            float* __restrict__ distsum) {
  int idx = blockIdx.x * blockDim.x + threadIdx.x;
  if (idx == 0) *distsum = 0.0f;
  if (idx < HH * DD) {                 // W1t[h][k] = W1[k][h]  (first 256 rows of W1 only)
    int h = idx / DD, k = idx % DD;
    W1t[h * DD + k] = f2bf(W1[(size_t)k * HH + h]);
  }
  if (idx < 16 * HH) {                 // W2t[e][k] = W2[k][e], zero-padded e=8..15
    int e = idx / HH, k = idx % HH;
    W2t[e * HH + k] = (e < EE) ? f2bf(W2[(size_t)k * EE + e]) : (unsigned short)0;
  }
}

// ---------------------------------------------------------------------------
// Kernel 1: global sum of all B*N*E Euclidean distances (for dists.mean())
// ---------------------------------------------------------------------------
__global__ void distsum_kernel(const float* __restrict__ xyz, const float* __restrict__ centers,
                               float* __restrict__ distsum) {
  __shared__ float sc[EE * 3];
  __shared__ float red[256];
  if (threadIdx.x < EE * 3) sc[threadIdx.x] = centers[threadIdx.x];
  __syncthreads();
  float acc = 0.0f;
  const int total = BB * NN;
  for (int i = blockIdx.x * blockDim.x + threadIdx.x; i < total; i += gridDim.x * blockDim.x) {
    float x = xyz[3 * (size_t)i + 0];
    float y = xyz[3 * (size_t)i + 1];
    float z = xyz[3 * (size_t)i + 2];
#pragma unroll
    for (int e = 0; e < EE; ++e) {
      float dx = x - sc[3 * e + 0], dy = y - sc[3 * e + 1], dz = z - sc[3 * e + 2];
      acc += sqrtf(fmaxf(dx * dx + dy * dy + dz * dz, 0.0f));
    }
  }
  red[threadIdx.x] = acc;
  __syncthreads();
  for (int s = 128; s > 0; s >>= 1) {
    if ((int)threadIdx.x < s) red[threadIdx.x] += red[threadIdx.x + s];
    __syncthreads();
  }
  if (threadIdx.x == 0) atomicAdd(distsum, red[0]);
}

// ---------------------------------------------------------------------------
// Kernel 2: fused gate MLP (WMMA bf16) + spatial affinity -> logits (B,E,N)
//   Block = 256 threads = 8 waves, handles 128 tokens (one 16-token tile
//   per pair of... no: each wave owns one 16-token tile? TPB/16 = 4 tiles,
//   waves 2w and 2w+1? -> Simpler: 8 waves, waves w and w+4 share nothing;
//   here: wave w owns tile (w & 3), two waves per tile would collide, so we
//   use 4 tiles and 8 waves with wave w owning tile w%4 only for w<4?  No —
//   we give each wave its OWN tile: TPB = 64 needs 4 waves; with 8 waves we
//   let waves 4..7 own tiles from the next 64 tokens. So a block covers 128
//   tokens = 8 independent 16-token tiles, one per wave.
//   Per wave: 8 h-slices x 8 WMMA (GEMM1, K=256) + GELU -> private LDS h
//   tile -> 4 WMMA (GEMM2, K=128, experts padded to 16) -> logits. No
//   cross-wave barrier after staging (same-wave DS ops are in-order).
//   W1^T (bf16, 64KB) is staged global->LDS once per block by the Tensor
//   Data Mover (tensor_load_to_lds + s_wait_tensorcnt) when available.
// ---------------------------------------------------------------------------
#define TOKS_PER_BLOCK (16 * 8)   // 8 waves x 16-token tile = 128 tokens

__global__ __launch_bounds__(256)
void gate_gemm_kernel(const float* __restrict__ tokens, const float* __restrict__ xyz,
                      const float* __restrict__ W1, const float* __restrict__ b1,
                      const float* __restrict__ b2, const float* __restrict__ centers,
                      const unsigned short* __restrict__ W1t_u,
                      const unsigned short* __restrict__ W2t_u,
                      const float* __restrict__ distsum, float* __restrict__ logits) {
  const __bf16* W2t = (const __bf16*)W2t_u;

  __shared__ __align__(32) __bf16 sW1[HH * DD];              // 64 KB: W1^T via TDM
  __shared__ __align__(32) __bf16 sA[TOKS_PER_BLOCK][DD];    // 64 KB: token tile bf16
  __shared__ __align__(32) __bf16 sH[8][16][HH];             // 32 KB: per-wave h tiles
  __shared__ float sXYZ[TOKS_PER_BLOCK][3];                  // 1.5 KB

  const int b  = blockIdx.x / (NN / TOKS_PER_BLOCK);
  const int n0 = (blockIdx.x % (NN / TOKS_PER_BLOCK)) * TOKS_PER_BLOCK;

#if HAVE_TDM
  if (threadIdx.x < 32) {  // wave 0 issues one TDM descriptor: 1x32768 bf16 row
    unsigned           lds_off = (unsigned)(size_t)&sW1[0];     // low 32 bits = LDS byte offset
    unsigned long long ga      = (unsigned long long)(size_t)W1t_u;
    tdm_g0_t g0;
    g0[0] = 1u;                                                // count=1, user descriptor
    g0[1] = lds_off;                                           // lds_addr
    g0[2] = (unsigned)(ga & 0xFFFFFFFFull);                    // global_addr[31:0]
    g0[3] = (unsigned)((ga >> 32) & 0x01FFFFFFull) | (2u << 30); // global_addr[56:32] | type=2
    tdm_g1_t g1;
    g1[0] = 0x00010000;        // workgroup_mask=0, data_size=1 (2 bytes)
    g1[1] = (int)0x80000000u;  // tensor_dim0 = 32768 (bits 79:48, low half)
    g1[2] = 0x00010000;        // tensor_dim0 hi = 0, tensor_dim1 = 1
    g1[3] = (int)0x80000000u;  // tensor_dim1 hi = 0, tile_dim0 = 32768
    g1[4] = 1;                 // tile_dim1 = 1, tile_dim2 = 0
    g1[5] = 32768;             // tensor_dim0_stride[31:0]
    g1[6] = 0;                 // stride hi, tensor_dim1_stride lo
    g1[7] = 0;
    tdm_g2_t gz = {};
#if __clang_major__ >= 23
    tdm_g1_t z8 = {};
    __builtin_amdgcn_tensor_load_to_lds(g0, g1, gz, gz, z8, 0);
#else
    __builtin_amdgcn_tensor_load_to_lds(g0, g1, gz, gz, 0);
#endif
    __builtin_amdgcn_s_wait_tensorcnt(0);
  }
#else
  for (int i = threadIdx.x * 8; i < HH * DD; i += 256 * 8)
    *(v8bf*)&sW1[i] = *(const v8bf*)((const __bf16*)W1t_u + i);
#endif

  // Stage token tile f32 -> bf16 (vectorized b128 reads, b64 LDS writes)
  const float* tb = tokens + ((size_t)b * NN + n0) * DD;
  for (int i = threadIdx.x; i < TOKS_PER_BLOCK * DD / 4; i += 256) {
    const float4 v = ((const float4*)tb)[i];
    __bf16* dst = &sA[0][0] + (size_t)i * 4;
    dst[0] = (__bf16)v.x; dst[1] = (__bf16)v.y; dst[2] = (__bf16)v.z; dst[3] = (__bf16)v.w;
  }
  for (int i = threadIdx.x; i < TOKS_PER_BLOCK * 3; i += 256)
    (&sXYZ[0][0])[i] = xyz[((size_t)b * NN + n0) * 3 + i];
  __syncthreads();   // sW1 (TDM done by wave0 before barrier) + sA + sXYZ ready

  const int lane    = threadIdx.x & 31;   // wave32
  const int wv      = threadIdx.x >> 5;   // wave owns token rows [wv*16, wv*16+16)
  const int col     = lane & 15;
  const int hi      = lane >> 4;
  const int m_a     = lane & 15;
  const int rowbase = wv * 16;

  // ---- GEMM1: all 8 h-slices for this wave's 16 tokens -------------------
#pragma unroll
  for (int s = 0; s < 8; ++s) {
    const int hcol = s * 16 + col;
    // accumulator init = b1 + xyz-tail (gate_input columns 256..258)
    v8f acc;
#pragma unroll
    for (int r = 0; r < 8; ++r) {
      int m = rowbase + r + hi * 8;
      float v = b1[hcol];
      v += sXYZ[m][0] * W1[(size_t)(DD + 0) * HH + hcol];
      v += sXYZ[m][1] * W1[(size_t)(DD + 1) * HH + hcol];
      v += sXYZ[m][2] * W1[(size_t)(DD + 2) * HH + hcol];
      acc[r] = v;
    }
#pragma unroll
    for (int kb = 0; kb < DD; kb += 32) {
      // A 16x32 bf16 fragment: lanes 0-15 K=0..15, lanes 16-31 K=16..31
      int ko = kb + hi * 16;
      v8bf alo = *(const v8bf*)&sA[rowbase + m_a][ko];
      v8bf ahi = *(const v8bf*)&sA[rowbase + m_a][ko + 8];
      v16bf afrag;
#pragma unroll
      for (int i = 0; i < 8; ++i) { afrag[i] = alo[i]; afrag[8 + i] = ahi[i]; }
      // B 32x16 bf16 fragment from LDS (contiguous 32 B per lane)
      v16bf bfrag = *(const v16bf*)&sW1[hcol * DD + kb + hi * 16];
      acc = __builtin_amdgcn_wmma_f32_16x16x32_bf16(false, afrag, false, bfrag,
                                                    (short)0, acc, false, false);
    }
    // exact-erf GELU -> per-wave private h tile (same-wave DS ops stay in order)
#pragma unroll
    for (int r = 0; r < 8; ++r) {
      float x = acc[r];
      sH[wv][r + hi * 8][hcol] = (__bf16)(0.5f * x * (1.0f + erff(x * 0.70710678118654752f)));
    }
  }

  // ---- GEMM2: 16x16x128 (experts padded to 16 columns), 4 WMMA steps -----
  v8f acc2;
#pragma unroll
  for (int r = 0; r < 8; ++r) acc2[r] = 0.0f;
#pragma unroll
  for (int kb = 0; kb < HH; kb += 32) {
    int ko = kb + hi * 16;
    v8bf alo = *(const v8bf*)&sH[wv][m_a][ko];
    v8bf ahi = *(const v8bf*)&sH[wv][m_a][ko + 8];
    v16bf afrag;
#pragma unroll
    for (int i = 0; i < 8; ++i) { afrag[i] = alo[i]; afrag[8 + i] = ahi[i]; }
    v16bf bfrag = *(const v16bf*)(W2t + (size_t)col * HH + kb + hi * 16);
    acc2 = __builtin_amdgcn_wmma_f32_16x16x32_bf16(false, afrag, false, bfrag,
                                                   (short)0, acc2, false, false);
  }

  // ---- logits = content + bias + spatial affinity, stored (B,E,N) --------
  const float inv_den = 1.0f / ((*distsum) * (1.0f / (float)(BB * NN * EE)) + 1e-6f);
  const int e = col;
  if (e < EE) {
#pragma unroll
    for (int r = 0; r < 8; ++r) {
      int m   = rowbase + r + hi * 8;
      int tok = n0 + m;
      float dx = sXYZ[m][0] - centers[3 * e + 0];
      float dy = sXYZ[m][1] - centers[3 * e + 1];
      float dz = sXYZ[m][2] - centers[3 * e + 2];
      float dist = sqrtf(fmaxf(dx * dx + dy * dy + dz * dz, 0.0f));
      logits[((size_t)(b * EE + e)) * NN + tok] = acc2[r] + b2[e] - dist * inv_den;
    }
  }
}

// ---------------------------------------------------------------------------
// Kernel 3: exact k-th-largest key per (b,e) row via 4x8-bit radix select
// ---------------------------------------------------------------------------
__global__ void select_kth_kernel(const float* __restrict__ logits, unsigned* __restrict__ thresh) {
  const int row = blockIdx.x;                      // b*E + e
  const float* p = logits + (size_t)row * NN;
  __shared__ unsigned hist[256];
  __shared__ unsigned s_prefix;
  __shared__ int s_k;
  if (threadIdx.x == 0) { s_prefix = 0u; s_k = KSEL; }
  __syncthreads();
  for (int pass = 0; pass < 4; ++pass) {
    int shift = 24 - 8 * pass;
    hist[threadIdx.x] = 0u;
    __syncthreads();
    unsigned mask = (pass == 0) ? 0u : (0xFFFFFFFFu << (shift + 8));
    unsigned pref = s_prefix;
    for (int i = threadIdx.x; i < NN; i += 256) {
      unsigned key = fkey(p[i]);
      if ((key & mask) == pref)
        atomicAdd(&hist[(key >> shift) & 255u], 1u);
    }
    __syncthreads();
    if (threadIdx.x == 0) {
      int kk = s_k, bin = 0;
      for (int j = 255; j >= 0; --j) {
        int c = (int)hist[j];
        if (kk <= c) { bin = j; break; }
        kk -= c;
      }
      s_k = kk;
      s_prefix |= ((unsigned)bin << shift);
    }
    __syncthreads();
  }
  if (threadIdx.x == 0) thresh[row] = s_prefix;    // exact key of the k-th largest
}

// ---------------------------------------------------------------------------
// Kernel 4: per-token dispatch/floor/cap/combine; writes both outputs
// ---------------------------------------------------------------------------
__global__ void finalize_kernel(const float* __restrict__ logits, const unsigned* __restrict__ thresh,
                                const int* __restrict__ t, float* __restrict__ out) {
  const int blocksPerB = NN / 256;
  const int b = blockIdx.x / blocksPerB;
  const int n = (blockIdx.x % blocksPerB) * 256 + threadIdx.x;
  __shared__ unsigned sth[EE];
  __shared__ float scap;
  if (threadIdx.x < EE) sth[threadIdx.x] = thresh[b * EE + threadIdx.x];
  if (threadIdx.x == 0) scap = 0.5f + 1.1f * ((float)t[b] / 1000.0f);  // CAP_LOW + (CAP_HIGH+CAP_LOW)*t/T_MAX
  __syncthreads();

  float d[EE], hr[EE];
  const float cap = scap;
  float exsum = 0.0f, hrsum = 0.0f;
#pragma unroll
  for (int e = 0; e < EE; ++e) {
    float lg = logits[((size_t)(b * EE + e)) * NN + n];
    unsigned key = fkey(lg);
    float v = (key >= sth[e]) ? (1.0f / (1.0f + expf(-lg))) : 0.0f;  // sigmoid on top-k
    v = 0.7f * v + 0.0375f;                    // floor: (1-alpha)*d + alpha/E, alpha=0.3
    float ex = fmaxf(v - cap, 0.0f);
    float c  = v - ex;
    float h  = fmaxf(cap - c, 0.0f);
    d[e] = c; hr[e] = h;
    exsum += ex; hrsum += h;
  }
  hrsum = fmaxf(hrsum, 1e-8f);
  float sum = 0.0f;
#pragma unroll
  for (int e = 0; e < EE; ++e) {
    d[e] += exsum * hr[e] / hrsum;             // proportional redistribution
    sum += d[e];
  }
  float inv = 1.0f / (sum + 1e-8f);
  size_t base = ((size_t)b * NN + n) * EE;
#pragma unroll
  for (int e = 0; e < EE; ++e) {
    out[base + e] = d[e];                                   // dispatch
    out[(size_t)BB * NN * EE + base + e] = d[e] * inv;      // combine
  }
}

// ---------------------------------------------------------------------------
extern "C" void kernel_launch(void* const* d_in, const int* in_sizes, int n_in,
                              void* d_out, int out_size, void* d_ws, size_t ws_size,
                              hipStream_t stream) {
  (void)in_sizes; (void)n_in; (void)out_size; (void)ws_size;
  const float* tokens  = (const float*)d_in[0];
  const float* xyz     = (const float*)d_in[1];
  const float* W1      = (const float*)d_in[2];
  const float* b1      = (const float*)d_in[3];
  const float* W2      = (const float*)d_in[4];
  const float* b2      = (const float*)d_in[5];
  const float* centers = (const float*)d_in[6];
  const int*   t       = (const int*)d_in[7];
  float* out = (float*)d_out;

  char* ws = (char*)d_ws;
  float*          distsum = (float*)ws;                                  // [0,64)
  unsigned short* W1t     = (unsigned short*)(ws + 64);                  // 128*256*2 = 65536 B
  unsigned short* W2t     = (unsigned short*)(ws + 64 + 65536);          // 16*128*2  = 4096 B
  float*          logits  = (float*)(ws + 64 + 65536 + 4096);            // B*E*N*4   = 4 MB
  unsigned*       thresh  = (unsigned*)(ws + 64 + 65536 + 4096 + (size_t)BB * EE * NN * 4);

  prep_kernel<<<128, 256, 0, stream>>>(W1, W2, W1t, W2t, distsum);
  distsum_kernel<<<512, 256, 0, stream>>>(xyz, centers, distsum);
  gate_gemm_kernel<<<(BB * NN) / TOKS_PER_BLOCK, 256, 0, stream>>>(tokens, xyz, W1, b1, b2,
                                                                   centers, W1t, W2t, distsum,
                                                                   logits);
  select_kth_kernel<<<BB * EE, 256, 0, stream>>>(logits, thresh);
  finalize_kernel<<<BB * (NN / 256), 256, 0, stream>>>(logits, thresh, t, out);
}
